// MoE_10136122819137
// MI455X (gfx1250) — matrix operand order repaired and checked by reading, not compiled, not run on hardware
//
#include <hip/hip_runtime.h>
#include <cstddef>

// ---------------- problem constants ----------------
constexpr int D    = 1024;   // model dim
constexpr int NE   = 16;     // experts
constexpr int TOPK = 2;
constexpr int I_R  = 1408;   // routed intermediate
constexpr int SI   = 2816;   // shared intermediate
constexpr int T    = 4096;   // tokens (2*2048)
constexpr int BM   = 128;    // block tile M
constexpr int BN   = 128;    // block tile N
constexpr int BK   = 32;     // k-chunk (one bf16 WMMA K)
constexpr int TILES = 32;    // max 128-row tiles per expert (4096/128)
constexpr int MAXROWS = 10240; // 8192 assignments + 16*127 padding, rounded up

typedef __bf16 bf16;
typedef __bf16 v16bf __attribute__((ext_vector_type(16)));
typedef float  v8f   __attribute__((ext_vector_type(8)));
typedef int    v4i   __attribute__((ext_vector_type(4)));

struct __attribute__((aligned(16))) U4 { unsigned int x, y, z, w; };
struct __attribute__((aligned(16))) F4 { float x, y, z, w; };
struct __attribute__((aligned(8)))  H4 { bf16 a, b, c, d; };

union Frag { struct { U4 lo, hi; } u; v16bf v; };

// ---------------- CDNA5 async global->LDS copy (guarded) ----------------
#if defined(__HIP_DEVICE_COMPILE__) && __has_builtin(__builtin_amdgcn_global_load_async_to_lds_b128)
#define HAVE_ASYNC_LDS 1
#else
#define HAVE_ASYNC_LDS 0
#endif

__device__ __forceinline__ void copy16_g2lds(const bf16* g, bf16* l) {
#if HAVE_ASYNC_LDS
  __builtin_amdgcn_global_load_async_to_lds_b128(
      (__attribute__((address_space(1))) v4i*)g,
      (__attribute__((address_space(3))) v4i*)l, 0, 0);
#else
  *(U4*)l = *(const U4*)g;
#endif
}

__device__ __forceinline__ void async_drain() {
#if HAVE_ASYNC_LDS
#if __has_builtin(__builtin_amdgcn_s_wait_asynccnt)
  __builtin_amdgcn_s_wait_asynccnt(0);
#else
  asm volatile("s_wait_asynccnt 0" ::: "memory");
#endif
#endif
}

// ---------------- small kernels ----------------

__global__ void init_meta_kernel(int* counts, int* cursors) {
  int i = threadIdx.x;
  if (i < NE) { counts[i] = 0; cursors[i] = 0; }
}

__global__ void init_rows_kernel(int* row_tok, float* row_w) {
  int i = blockIdx.x * blockDim.x + threadIdx.x;
  if (i < MAXROWS) { row_tok[i] = 0; row_w[i] = 0.f; }
}

// out[t*D+n] = sb2[n]  (folds the shared-expert output bias in once)
__global__ void init_out_kernel(float* __restrict__ out, const float* __restrict__ sb2) {
  int i = blockIdx.x * blockDim.x + threadIdx.x;
  if (i < T * D) out[i] = sb2[i & (D - 1)];
}

__global__ void cvt_bf16_kernel(const float* __restrict__ src, bf16* __restrict__ dst, long n) {
  long i = ((long)blockIdx.x * blockDim.x + threadIdx.x) * 4;
  if (i + 3 < n) {
    F4 v = *(const F4*)(src + i);
    H4 h;
    h.a = (bf16)v.x; h.b = (bf16)v.y; h.c = (bf16)v.z; h.d = (bf16)v.w;
    *(H4*)(dst + i) = h;
  } else {
    for (; i < n; ++i) dst[i] = (bf16)src[i];
  }
}

// Gate: logits = x @ gate_w + gate_b ; softmax(fp32) ; top-2 ; count per expert.
__global__ void gate_kernel(const float* __restrict__ x, const float* __restrict__ gw,
                            const float* __restrict__ gb, int* __restrict__ counts,
                            int* __restrict__ top_idx, float* __restrict__ top_w) {
  int t = blockIdx.x * blockDim.x + threadIdx.x;
  if (t >= T) return;
  float acc[NE];
#pragma unroll
  for (int e = 0; e < NE; ++e) acc[e] = gb[e];
  const float* xr = x + (size_t)t * D;
  for (int d = 0; d < D; ++d) {
    float xv = xr[d];
    const float* gr = gw + (size_t)d * NE;
#pragma unroll
    for (int e = 0; e < NE; ++e) acc[e] = fmaf(xv, gr[e], acc[e]);
  }
  float mx = acc[0];
#pragma unroll
  for (int e = 1; e < NE; ++e) mx = fmaxf(mx, acc[e]);
  float sum = 0.f;
#pragma unroll
  for (int e = 0; e < NE; ++e) { acc[e] = __expf(acc[e] - mx); sum += acc[e]; }
  float inv = 1.f / sum;
  int i0 = 0; float v0 = -1.f;
#pragma unroll
  for (int e = 0; e < NE; ++e) if (acc[e] > v0) { v0 = acc[e]; i0 = e; }
  int i1 = 0; float v1 = -1.f;
#pragma unroll
  for (int e = 0; e < NE; ++e) if (e != i0 && acc[e] > v1) { v1 = acc[e]; i1 = e; }
  top_idx[t * 2 + 0] = i0;  top_w[t * 2 + 0] = v0 * inv;  // ROUTE_SCALE == 1.0
  top_idx[t * 2 + 1] = i1;  top_w[t * 2 + 1] = v1 * inv;
  atomicAdd(&counts[i0], 1);
  atomicAdd(&counts[i1], 1);
}

// padded exclusive scan of counts (16 entries): each expert bucket rounded up to 128 rows
__global__ void offsets_kernel(const int* __restrict__ counts, int* __restrict__ padded_off) {
  if (threadIdx.x == 0 && blockIdx.x == 0) {
    int o = 0;
    for (int e = 0; e < NE; ++e) {
      padded_off[e] = o;
      o += ((counts[e] + BM - 1) / BM) * BM;
    }
    padded_off[NE] = o;
  }
}

__global__ void scatter_kernel(const int* __restrict__ top_idx, const float* __restrict__ top_w,
                               const int* __restrict__ padded_off, int* __restrict__ cursors,
                               int* __restrict__ row_tok, float* __restrict__ row_w) {
  int t = blockIdx.x * blockDim.x + threadIdx.x;
  if (t >= T) return;
#pragma unroll
  for (int k = 0; k < TOPK; ++k) {
    int e = top_idx[t * 2 + k];
    float w = top_w[t * 2 + k];
    int slot = atomicAdd(&cursors[e], 1);
    int row = padded_off[e] + slot;
    row_tok[row] = t;
    row_w[row] = w;
  }
}

// ---------------- WMMA GEMM kernels ----------------
// Block: 256 threads = 8 waves, 4 waves along M x 2 along N.
// Each wave: 32x64 region = 2(M) x 4(N) 16x16 tiles.

// up-proj: act = silu(x@W1 + b1) * (x@W3 + b3), written as bf16 rows of the bucket.
__global__ __launch_bounds__(256)
void ffn_up_kernel(const bf16* __restrict__ xb, const bf16* __restrict__ w1,
                   const bf16* __restrict__ w3, const float* __restrict__ b1,
                   const float* __restrict__ b3, bf16* __restrict__ act,
                   const int* __restrict__ row_tok, const int* __restrict__ counts,
                   const int* __restrict__ padded_off, int tiles_per_e, int I) {
  const int e    = blockIdx.x / tiles_per_e;
  const int tile = blockIdx.x % tiles_per_e;
  int rowbase;
  if (counts) {
    int count = counts[e];
    if (tile * BM >= count) return;
    rowbase = padded_off[e] + tile * BM;
  } else {
    rowbase = tile * BM;       // identity (shared expert), grid sized exactly
  }
  const int n0 = blockIdx.y * BN;

  __shared__ bf16 ldsA [BM * BK];
  __shared__ bf16 ldsB1[BN * BK];   // stored transposed: [n][k]
  __shared__ bf16 ldsB3[BN * BK];
  __shared__ int  ldsTok[BM];

  const int tid  = threadIdx.x;
  const int lane = tid & 31;
  const int wave = tid >> 5;
  const int waveM = wave & 3;
  const int waveN = wave >> 2;

  if (tid < BM) ldsTok[tid] = row_tok ? row_tok[rowbase + tid] : (rowbase + tid);
  __syncthreads();

  const size_t we = (size_t)e * (size_t)D * (size_t)I;
  const bf16* w1e = w1 + we;
  const bf16* w3e = w3 + we;

  // B transpose task: two K rows per thread, packed b32 LDS stores
  const int bk2 = (tid >> 4) * 2;    // even k: 0,2,..,30
  const int bnc = tid & 15;          // chunk of 8 n

  v8f hacc[2][4] = {};
  v8f gacc[2][4] = {};

  for (int kc = 0; kc < D; kc += BK) {
    // A tile: 128x32 bf16, row major, 512 x 16B async chunks
#pragma unroll
    for (int it = 0; it < 2; ++it) {
      int idx = tid + it * 256;      // 0..511
      int r = idx >> 2;
      int seg = idx & 3;
      int tok = ldsTok[r];
      copy16_g2lds(xb + (size_t)tok * D + kc + seg * 8, ldsA + idx * 8);
    }
    // B tiles: global [K][N] -> LDS [N][K], (k,k+1) pairs packed into b32 stores
    {
      size_t g0 = (size_t)(kc + bk2) * I + n0 + bnc * 8;
      U4 a1 = *(const U4*)(w1e + g0);
      U4 c1 = *(const U4*)(w1e + g0 + I);
      U4 a3 = *(const U4*)(w3e + g0);
      U4 c3 = *(const U4*)(w3e + g0 + I);
      if (kc + BK < D) {  // prime L2/WGP$ for the next k-chunk (global_prefetch_b8)
        __builtin_prefetch(w1e + g0 + (size_t)BK * I, 0, 0);
        __builtin_prefetch(w3e + g0 + (size_t)BK * I, 0, 0);
      }
      const unsigned short* pa1 = (const unsigned short*)&a1;
      const unsigned short* pc1 = (const unsigned short*)&c1;
      const unsigned short* pa3 = (const unsigned short*)&a3;
      const unsigned short* pc3 = (const unsigned short*)&c3;
#pragma unroll
      for (int j = 0; j < 8; ++j) {
        int n = bnc * 8 + j;
        *(unsigned*)(ldsB1 + n * BK + bk2) = (unsigned)pa1[j] | ((unsigned)pc1[j] << 16);
        *(unsigned*)(ldsB3 + n * BK + bk2) = (unsigned)pa3[j] | ((unsigned)pc3[j] << 16);
      }
    }
    async_drain();
    __syncthreads();

    // A fragments: lane<16 -> K {0..7, 16..23}; lane>=16 -> K {8..15, 24..31}
    Frag af[2];
    const int akoff = (lane & 16) ? 8 : 0;
#pragma unroll
    for (int mi = 0; mi < 2; ++mi) {
      int r = waveM * 32 + mi * 16 + (lane & 15);
      const U4* pa = (const U4*)(ldsA + r * BK + akoff);
      af[mi].u.lo = pa[0];
      af[mi].u.hi = pa[2];   // +16 bf16
    }
    const int bkoff = (lane & 16) ? 8 : 0;
#pragma unroll
    for (int ni = 0; ni < 4; ++ni) {
      int n = waveN * 64 + ni * 16 + (lane & 15);
      const U4* pb1 = (const U4*)(ldsB1 + n * BK + bkoff);
      const U4* pb3 = (const U4*)(ldsB3 + n * BK + bkoff);
      Frag f1, f3;
      f1.u.lo = pb1[0]; f1.u.hi = pb1[2];
      f3.u.lo = pb3[0]; f3.u.hi = pb3[2];
#pragma unroll
      for (int mi = 0; mi < 2; ++mi) {
        hacc[mi][ni] = __builtin_amdgcn_wmma_f32_16x16x32_bf16(
            false, af[mi].v, false, f1.v, (short)0, hacc[mi][ni], false, false);
        gacc[mi][ni] = __builtin_amdgcn_wmma_f32_16x16x32_bf16(
            false, af[mi].v, false, f3.v, (short)0, gacc[mi][ni], false, false);
      }
    }
    __syncthreads();
  }

  // epilogue: bias + SwiGLU -> bf16 activation rows
#pragma unroll
  for (int ni = 0; ni < 4; ++ni) {
    int n = n0 + waveN * 64 + ni * 16 + (lane & 15);
    float b1v = b1[(size_t)e * I + n];
    float b3v = b3[(size_t)e * I + n];
#pragma unroll
    for (int mi = 0; mi < 2; ++mi) {
      int mbase = waveM * 32 + mi * 16 + ((lane & 16) ? 8 : 0);
#pragma unroll
      for (int r = 0; r < 8; ++r) {
        float h = hacc[mi][ni][r] + b1v;
        float g = gacc[mi][ni][r] + b3v;
        float a = (h / (1.f + __expf(-h))) * g;
        act[(size_t)(rowbase + mbase + r) * (size_t)I + n] = (bf16)a;
      }
    }
  }
}

// down-proj: out[tok] += row_w * (act @ W2 + b2)  via fp32 global atomics
__global__ __launch_bounds__(256)
void ffn_down_kernel(const bf16* __restrict__ act, const bf16* __restrict__ w2,
                     const float* __restrict__ b2, float* __restrict__ out,
                     const int* __restrict__ row_tok, const float* __restrict__ row_w,
                     const int* __restrict__ counts, const int* __restrict__ padded_off,
                     int tiles_per_e, int I) {
  const int e    = blockIdx.x / tiles_per_e;
  const int tile = blockIdx.x % tiles_per_e;
  int rowbase;
  if (counts) {
    int count = counts[e];
    if (tile * BM >= count) return;
    rowbase = padded_off[e] + tile * BM;
  } else {
    rowbase = tile * BM;
  }
  const int n0 = blockIdx.y * BN;

  __shared__ bf16 ldsA[BM * BK];
  __shared__ bf16 ldsB[BN * BK];    // transposed [n][k]
  __shared__ int   ldsTok[BM];
  __shared__ float ldsW[BM];

  const int tid  = threadIdx.x;
  const int lane = tid & 31;
  const int wave = tid >> 5;
  const int waveM = wave & 3;
  const int waveN = wave >> 2;

  if (tid < BM) {
    int gr = rowbase + tid;
    ldsTok[tid] = row_tok ? row_tok[gr] : gr;
    ldsW[tid]   = row_w ? row_w[gr] : 1.f;
  }

  const bf16* w2e = w2 + (size_t)e * (size_t)I * (size_t)D;
  const int bk2 = (tid >> 4) * 2;
  const int bnc = tid & 15;

  v8f acc[2][4] = {};

  for (int kc = 0; kc < I; kc += BK) {
#pragma unroll
    for (int it = 0; it < 2; ++it) {
      int idx = tid + it * 256;
      int r = idx >> 2;
      int seg = idx & 3;
      copy16_g2lds(act + (size_t)(rowbase + r) * (size_t)I + kc + seg * 8, ldsA + idx * 8);
    }
    {
      size_t g0 = (size_t)(kc + bk2) * D + n0 + bnc * 8;
      U4 a = *(const U4*)(w2e + g0);
      U4 c = *(const U4*)(w2e + g0 + D);
      if (kc + BK < I) __builtin_prefetch(w2e + g0 + (size_t)BK * D, 0, 0);
      const unsigned short* pa = (const unsigned short*)&a;
      const unsigned short* pc = (const unsigned short*)&c;
#pragma unroll
      for (int j = 0; j < 8; ++j) {
        int n = bnc * 8 + j;
        *(unsigned*)(ldsB + n * BK + bk2) = (unsigned)pa[j] | ((unsigned)pc[j] << 16);
      }
    }
    async_drain();
    __syncthreads();

    Frag af[2];
    const int akoff = (lane & 16) ? 8 : 0;
#pragma unroll
    for (int mi = 0; mi < 2; ++mi) {
      int r = waveM * 32 + mi * 16 + (lane & 15);
      const U4* pa = (const U4*)(ldsA + r * BK + akoff);
      af[mi].u.lo = pa[0];
      af[mi].u.hi = pa[2];
    }
    const int bkoff = (lane & 16) ? 8 : 0;
#pragma unroll
    for (int ni = 0; ni < 4; ++ni) {
      int n = waveN * 64 + ni * 16 + (lane & 15);
      const U4* pb = (const U4*)(ldsB + n * BK + bkoff);
      Frag fb;
      fb.u.lo = pb[0]; fb.u.hi = pb[2];
#pragma unroll
      for (int mi = 0; mi < 2; ++mi) {
        acc[mi][ni] = __builtin_amdgcn_wmma_f32_16x16x32_bf16(
            false, af[mi].v, false, fb.v, (short)0, acc[mi][ni], false, false);
      }
    }
    __syncthreads();
  }

#pragma unroll
  for (int ni = 0; ni < 4; ++ni) {
    int n = n0 + waveN * 64 + ni * 16 + (lane & 15);
    float b2v = b2 ? b2[(size_t)e * D + n] : 0.f;
#pragma unroll
    for (int mi = 0; mi < 2; ++mi) {
      int mbase = waveM * 32 + mi * 16 + ((lane & 16) ? 8 : 0);
#pragma unroll
      for (int r = 0; r < 8; ++r) {
        int m = mbase + r;
        float val = (acc[mi][ni][r] + b2v) * ldsW[m];
        atomicAdd(out + (size_t)ldsTok[m] * D + n, val);
      }
    }
  }
}

// ---------------- host launch ----------------

extern "C" void kernel_launch(void* const* d_in, const int* in_sizes, int n_in,
                              void* d_out, int out_size, void* d_ws, size_t ws_size,
                              hipStream_t stream) {
  const float* x      = (const float*)d_in[0];
  const float* gate_w = (const float*)d_in[1];
  const float* gate_b = (const float*)d_in[2];
  const float* w1     = (const float*)d_in[3];
  const float* b1     = (const float*)d_in[4];
  const float* w3     = (const float*)d_in[5];
  const float* b3     = (const float*)d_in[6];
  const float* w2     = (const float*)d_in[7];
  const float* b2     = (const float*)d_in[8];
  const float* sw1    = (const float*)d_in[9];
  const float* sb1    = (const float*)d_in[10];
  const float* sw3    = (const float*)d_in[11];
  const float* sb3    = (const float*)d_in[12];
  const float* sw2    = (const float*)d_in[13];
  const float* sb2    = (const float*)d_in[14];
  float* out = (float*)d_out;

  char* ws = (char*)d_ws;
  size_t off = 0;
  auto carve = [&](size_t bytes) -> char* {
    char* p = ws + off;
    off = (off + bytes + 255) & ~(size_t)255;
    return p;
  };

  const size_t nX  = (size_t)T * D;
  const size_t nW1 = (size_t)NE * D * I_R;   // same for w3
  const size_t nW2 = (size_t)NE * I_R * D;
  const size_t nSW = (size_t)D * SI;         // sw1/sw3; sw2 is SI*D == same count

  bf16* xb   = (bf16*)carve(nX * 2);
  bf16* w1b  = (bf16*)carve(nW1 * 2);
  bf16* w3b  = (bf16*)carve(nW1 * 2);
  bf16* w2b  = (bf16*)carve(nW2 * 2);
  bf16* sw1b = (bf16*)carve(nSW * 2);
  bf16* sw3b = (bf16*)carve(nSW * 2);
  bf16* sw2b = (bf16*)carve(nSW * 2);
  bf16* actR = (bf16*)carve((size_t)MAXROWS * I_R * 2);
  bf16* actS = (bf16*)carve((size_t)T * SI * 2);
  int*   row_tok  = (int*)carve(MAXROWS * 4);
  float* row_w    = (float*)carve(MAXROWS * 4);
  int*   top_idx  = (int*)carve((size_t)T * 2 * 4);
  float* top_w    = (float*)carve((size_t)T * 2 * 4);
  int*   counts   = (int*)carve(NE * 4);
  int*   cursors  = (int*)carve(NE * 4);
  int*   padded   = (int*)carve((NE + 1) * 4);

  // ---- init ----
  init_meta_kernel<<<1, 32, 0, stream>>>(counts, cursors);
  init_rows_kernel<<<(MAXROWS + 255) / 256, 256, 0, stream>>>(row_tok, row_w);
  init_out_kernel<<<(T * D) / 256, 256, 0, stream>>>(out, sb2);

  // ---- fp32 -> bf16 conversions (weights become L2-resident in bf16) ----
  auto cvt = [&](const float* s, bf16* d, size_t n) {
    long blocks = (long)((n / 4 + 255) / 256) + 1;
    cvt_bf16_kernel<<<(unsigned)blocks, 256, 0, stream>>>(s, d, (long)n);
  };
  cvt(x,   xb,   nX);
  cvt(w1,  w1b,  nW1);
  cvt(w3,  w3b,  nW1);
  cvt(w2,  w2b,  nW2);
  cvt(sw1, sw1b, nSW);
  cvt(sw3, sw3b, nSW);
  cvt(sw2, sw2b, nSW);

  // ---- routing ----
  gate_kernel<<<T / 256, 256, 0, stream>>>(x, gate_w, gate_b, counts, top_idx, top_w);
  offsets_kernel<<<1, 1, 0, stream>>>(counts, padded);
  scatter_kernel<<<T / 256, 256, 0, stream>>>(top_idx, top_w, padded, cursors, row_tok, row_w);

  // ---- shared expert (identity routing, weight 1, sb2 folded into init_out) ----
  ffn_up_kernel<<<dim3(T / BM, SI / BN), 256, 0, stream>>>(
      xb, sw1b, sw3b, sb1, sb3, actS, nullptr, nullptr, nullptr, T / BM, SI);
  ffn_down_kernel<<<dim3(T / BM, D / BN), 256, 0, stream>>>(
      actS, sw2b, nullptr, out, nullptr, nullptr, nullptr, nullptr, T / BM, SI);

  // ---- routed experts ----
  ffn_up_kernel<<<dim3(NE * TILES, I_R / BN), 256, 0, stream>>>(
      xb, w1b, w3b, b1, b3, actR, row_tok, counts, padded, TILES, I_R);
  ffn_down_kernel<<<dim3(NE * TILES, D / BN), 256, 0, stream>>>(
      actR, w2b, b2, out, row_tok, row_w, counts, padded, TILES, I_R);

  (void)in_sizes; (void)n_in; (void)out_size; (void)ws_size;
}
